// EncoderLayer_58609123721699
// MI455X (gfx1250) — matrix operand order repaired and checked by reading, not compiled, not run on hardware
//
#include <hip/hip_runtime.h>
#include <hip/hip_bf16.h>
#include <math.h>

// ---------------------------------------------------------------------------
// Problem constants (from reference): B=2 N=256 D=512 E=128 H=4 NF=2048 EF=512
// Attention branch (qkv/rqk/scores/probs) is DEAD CODE in the reference.
// ---------------------------------------------------------------------------
#define Bc   2
#define Nc_  256
#define Dc   512
#define Ec   128
#define Hc   4
#define NFc  2048
#define EFc  512
#define ROWS_X (Bc * Nc_)          // 512
#define ROWS_P (Bc * Nc_ * Nc_)    // 131072

typedef __attribute__((ext_vector_type(16))) __bf16 v16bf;
typedef __attribute__((ext_vector_type(8)))  float  v8f;

__device__ __forceinline__ float gelu_f(float x) {
    return 0.5f * x * (1.0f + erff(x * 0.70710678118654752f));
}
__device__ __forceinline__ float sigm_f(float x) {
    return 1.0f / (1.0f + expf(-x));
}

// ---------------------------------------------------------------------------
// WMMA fragment helpers (CDNA5 v_wmma_f32_16x16x32_bf16, wave32).
// A (16x32 bf16) lane layout per ISA 7.12.2.
// ---------------------------------------------------------------------------
__device__ __forceinline__ v16bf load_a_frag(const __bf16* As, int lane,
                                             int kt, int stride) {
    int m  = lane & 15;
    int kh = (lane >> 4) << 3;               // 0 or 8
    const __bf16* row = As + m * stride + kt * 32;
    v16bf a;
#pragma unroll
    for (int v = 0; v < 4; ++v) {
        int k = kh + 2 * v;
        a[2 * v]     = row[k];
        a[2 * v + 1] = row[k + 1];
    }
#pragma unroll
    for (int v = 0; v < 4; ++v) {
        int k = 16 + kh + 2 * v;
        a[8 + 2 * v] = row[k];
        a[9 + 2 * v] = row[k + 1];
    }
    return a;
}

__device__ __forceinline__ v8f wmma_bf16(v16bf a, v16bf b, v8f c) {
    return __builtin_amdgcn_wmma_f32_16x16x32_bf16(
        false, a, false, b, (short)0, c, false, false);
}

// ---------------------------------------------------------------------------
// Weight conversion: f32 row-major (K x N) -> bf16, pre-swizzled so each
// lane's 16-element B fragment for tile (kt,nt) is one contiguous 32B load.
// ---------------------------------------------------------------------------
__global__ void k_wswz(const float* __restrict__ W, __bf16* __restrict__ out,
                       int K, int Ncols) {
    int tid = blockIdx.x * blockDim.x + threadIdx.x;
    int ntiles = Ncols >> 4;
    int total = (K >> 5) * ntiles * 32;
    if (tid >= total) return;
    int lane = tid & 31;
    int tile = tid >> 5;
    int n  = ((tile % ntiles) << 4) + (lane & 15);
    int kb = ((tile / ntiles) << 5) + ((lane >> 4) << 3);
    __bf16* o = out + (size_t)(tile * 32 + lane) * 16;
#pragma unroll
    for (int v = 0; v < 8; ++v) {
        o[v]     = (__bf16)W[(size_t)(kb + v) * Ncols + n];
        o[8 + v] = (__bf16)W[(size_t)(kb + 16 + v) * Ncols + n];
    }
}

// ---------------------------------------------------------------------------
// Generic row-LayerNorm (one block per row).
// ---------------------------------------------------------------------------
__global__ void k_ln(const float* __restrict__ X, const float* __restrict__ s,
                     const float* __restrict__ b, float* __restrict__ Y, int D) {
    __shared__ float red[256];
    int row = blockIdx.x, tid = threadIdx.x;
    const float* x = X + (size_t)row * D;
    float sm = 0.f, sq = 0.f;
    for (int e = tid; e < D; e += 256) { float v = x[e]; sm += v; sq += v * v; }
    red[tid] = sm; __syncthreads();
    for (int off = 128; off > 0; off >>= 1) { if (tid < off) red[tid] += red[tid + off]; __syncthreads(); }
    float mean = red[0] / (float)D; __syncthreads();
    red[tid] = sq; __syncthreads();
    for (int off = 128; off > 0; off >>= 1) { if (tid < off) red[tid] += red[tid + off]; __syncthreads(); }
    float var = red[0] / (float)D - mean * mean;
    float rstd = rsqrtf(var + 1e-5f);
    for (int e = tid; e < D; e += 256)
        Y[(size_t)row * D + e] = (x[e] - mean) * rstd * s[e] + b[e];
}

// ---------------------------------------------------------------------------
// Generic bf16-WMMA GEMM (x-FFN path): out = act(A @ Wsw + bias) (+ resid)
// 16 rows x 128 cols per block; K streamed through LDS in 512-wide chunks.
// ---------------------------------------------------------------------------
#define KC 512
__global__ void k_gemm(const float* __restrict__ A, const __bf16* __restrict__ Wsw,
                       const float* __restrict__ bias, const float* __restrict__ resid,
                       float* __restrict__ out, int K, int Ncols, int act) {
    __shared__ __bf16 As[16 * KC];
    int mt = blockIdx.x, ng = blockIdx.y;
    int tid = threadIdx.x, lane = tid & 31, wave = tid >> 5;
    int ntilesTotal = Ncols >> 4;
    int nt = ng * 8 + wave;
    const v16bf* Wt = (const v16bf*)Wsw;
    v8f c = {};
    for (int k0 = 0; k0 < K; k0 += KC) {
        int kc = (K - k0 < KC) ? (K - k0) : KC;
        __syncthreads();
        for (int idx = tid; idx < 16 * kc; idx += 256) {
            int r = idx / kc, k = idx - r * kc;
            As[r * KC + k] = (__bf16)A[(size_t)(mt * 16 + r) * K + k0 + k];
        }
        __syncthreads();
        int kt0 = k0 >> 5;
        for (int kk = 0; kk < (kc >> 5); ++kk) {
            v16bf a = load_a_frag(As, lane, kk, KC);
            v16bf b = Wt[((size_t)(kt0 + kk) * ntilesTotal + nt) * 32 + lane];
            c = wmma_bf16(a, b, c);
        }
    }
    int n = (nt << 4) + (lane & 15);
    int mb = (lane >> 4) << 3;
    float bv = bias[n];
#pragma unroll
    for (int r = 0; r < 8; ++r) {
        int row = mt * 16 + mb + r;
        float v = c[r] + bv;
        if (act == 1) v = gelu_f(v);
        size_t o = (size_t)row * Ncols + n;
        if (resid) v += resid[o];
        out[o] = v;
    }
}

// ---------------------------------------------------------------------------
// LN(x,en1) then o = xn @ o12_W + b  (512 -> 8), one block per row.
// ---------------------------------------------------------------------------
__global__ void k_ln_o(const float* __restrict__ X, const float* __restrict__ s,
                       const float* __restrict__ b, const float* __restrict__ W,
                       const float* __restrict__ bias, float* __restrict__ O) {
    __shared__ float xn[Dc];
    __shared__ float red[256];
    int row = blockIdx.x, tid = threadIdx.x;
    const float* xr = X + (size_t)row * Dc;
    float sm = 0.f, sq = 0.f;
    for (int e = tid; e < Dc; e += 256) { float v = xr[e]; sm += v; sq += v * v; }
    red[tid] = sm; __syncthreads();
    for (int off = 128; off > 0; off >>= 1) { if (tid < off) red[tid] += red[tid + off]; __syncthreads(); }
    float mean = red[0] / (float)Dc; __syncthreads();
    red[tid] = sq; __syncthreads();
    for (int off = 128; off > 0; off >>= 1) { if (tid < off) red[tid] += red[tid + off]; __syncthreads(); }
    float rstd = rsqrtf(red[0] / (float)Dc - mean * mean + 1e-5f);
    __syncthreads();
    for (int e = tid; e < Dc; e += 256) xn[e] = (xr[e] - mean) * rstd * s[e] + b[e];
    __syncthreads();
    int cidx = tid & 7, dg = tid >> 3;
    float acc = 0.f;
    for (int d = dg; d < Dc; d += 32) acc += xn[d] * W[d * 8 + cidx];
    red[tid] = acc; __syncthreads();
    for (int off = 128; off >= 8; off >>= 1) { if (tid < off) red[tid] += red[tid + off]; __syncthreads(); }
    if (tid < 8) O[(size_t)row * 8 + tid] = red[tid] + bias[tid];
}

// ---------------------------------------------------------------------------
// p1 = p + (o1 (x) o2) @ o3_W + o3_b
// ---------------------------------------------------------------------------
__global__ void k_pair1(const float* __restrict__ P, const float* __restrict__ O,
                        const float* __restrict__ o3W, const float* __restrict__ o3b,
                        float* __restrict__ Pout) {
    size_t idx = (size_t)blockIdx.x * 256 + threadIdx.x;
    int e = (int)(idx & 127);
    size_t r = idx >> 7;
    int bb = (int)(r >> 16);
    int rem = (int)(r & 65535);
    int i = rem >> 8, j = rem & 255;
    const float* o1 = O + ((size_t)(bb << 8) + i) * 8;
    const float* o2 = O + ((size_t)(bb << 8) + j) * 8 + 4;
    float acc = o3b[e];
#pragma unroll
    for (int h = 0; h < 4; ++h) {
        float a1 = o1[h];
#pragma unroll
        for (int g = 0; g < 4; ++g)
            acc += a1 * o2[g] * o3W[((h << 2) + g) * Ec + e];
    }
    Pout[idx] = P[idx] + acc;
}

__global__ void k_zero(float* p, int n) {
    int i = blockIdx.x * 256 + threadIdx.x;
    if (i < n) p[i] = 0.f;
}

// ---------------------------------------------------------------------------
// 64-row tile LayerNorm straight from global (read twice, L2-hot) -> bf16 LDS.
// 4 threads per row; 256 threads cover 64 rows in one pass.
// ---------------------------------------------------------------------------
__device__ __forceinline__ void tile_ln64(const float* __restrict__ Pg, size_t base,
                                          __bf16* As, float* red, float* mArr,
                                          float* rArr, const float* gs,
                                          const float* gb, int tid) {
    int rg = tid >> 2, tl = tid & 3;
    const float* row = Pg + base + (size_t)rg * Ec;
    float sm = 0.f, sq = 0.f;
    for (int e = tl; e < Ec; e += 4) { float v = row[e]; sm += v; sq += v * v; }
    red[tid] = sm; __syncthreads();
    if (tl < 2) red[tid] += red[tid + 2]; __syncthreads();
    if (tl < 1) red[tid] += red[tid + 1]; __syncthreads();
    if (tl == 0) mArr[rg] = red[tid] * (1.f / (float)Ec);
    __syncthreads();
    red[tid] = sq; __syncthreads();
    if (tl < 2) red[tid] += red[tid + 2]; __syncthreads();
    if (tl < 1) red[tid] += red[tid + 1]; __syncthreads();
    if (tl == 0) {
        float var = red[tid] * (1.f / (float)Ec) - mArr[rg] * mArr[rg];
        rArr[rg] = rsqrtf(var + 1e-5f);
    }
    __syncthreads();
    float mean = mArr[rg], rstd = rArr[rg];
    for (int e = tl; e < Ec; e += 4)
        As[rg * Ec + e] = (__bf16)((row[e] - mean) * rstd * gs[e] + gb[e]);
    __syncthreads();
}

// ---------------------------------------------------------------------------
// Fused triangle-gate producer, 64 rows (4 M-tiles) per block:
// pn=LN(p1,en2); t5=pn@t5_W+b (WMMA -> ws); tt=pn@t_W+b (WMMA);
// masked sigmoid reductions via atomics. Each B fragment loaded ONCE and
// reused by 4 WMMAs (4x less L2 weight traffic).
// ---------------------------------------------------------------------------
__global__ void k_pair_t5(const float* __restrict__ P1, const float* __restrict__ gs,
                          const float* __restrict__ gb, const __bf16* __restrict__ t5sw,
                          const float* __restrict__ t5b, const __bf16* __restrict__ tsw,
                          const float* __restrict__ tb, const int* __restrict__ mask,
                          float* __restrict__ t5out, float* __restrict__ abuf,
                          float* __restrict__ bbuf) {
    __shared__ __bf16 As[64 * Ec];     // 16 KB
    __shared__ float  Ts[64 * 16];     //  4 KB
    __shared__ float  red[256];
    __shared__ float  mArr[64], rArr[64];
    int mt = blockIdx.x, tid = threadIdx.x, lane = tid & 31, wave = tid >> 5;
    size_t base = (size_t)mt * 64 * Ec;
    tile_ln64(P1, base, As, red, mArr, rArr, gs, gb, tid);

    const v16bf* W5 = (const v16bf*)t5sw;   // 128x128, 8 ntiles
    const v16bf* Wt = (const v16bf*)tsw;    // 128x16,  1 ntile
    int nlo = lane & 15;
    int mb  = (lane >> 4) << 3;

    v8f c[4] = {};
#pragma unroll
    for (int kk = 0; kk < 4; ++kk) {
        v16bf b = W5[((size_t)kk * 8 + wave) * 32 + lane];
#pragma unroll
        for (int m = 0; m < 4; ++m) {
            v16bf a = load_a_frag(As + m * 16 * Ec, lane, kk, Ec);
            c[m] = wmma_bf16(a, b, c[m]);
        }
    }
    int n = (wave << 4) + nlo;
    float bv = t5b[n];
#pragma unroll
    for (int m = 0; m < 4; ++m)
#pragma unroll
        for (int r = 0; r < 8; ++r)
            t5out[((size_t)mt * 64 + m * 16 + mb + r) * Ec + n] = c[m][r] + bv;

    if (wave == 0) {
        v8f ct[4] = {};
#pragma unroll
        for (int kk = 0; kk < 4; ++kk) {
            v16bf b = Wt[(size_t)kk * 32 + lane];
#pragma unroll
            for (int m = 0; m < 4; ++m) {
                v16bf a = load_a_frag(As + m * 16 * Ec, lane, kk, Ec);
                ct[m] = wmma_bf16(a, b, ct[m]);
            }
        }
        float tbv = tb[nlo];
#pragma unroll
        for (int m = 0; m < 4; ++m)
#pragma unroll
            for (int r = 0; r < 8; ++r)
                Ts[(m * 16 + mb + r) * 16 + nlo] = ct[m][r] + tbv;
    }
    __syncthreads();

    {   // gate partial sums: 256 threads = 64 rows x 4 heads
        int mr = tid >> 2, h = tid & 3;
        size_t rowg = (size_t)mt * 64 + mr;
        int bb = (int)(rowg >> 16);
        int rem = (int)(rowg & 65535);
        int i = rem >> 8, j = rem & 255;
        int mk = mask[rowg];
        float t1 = Ts[mr * 16 + h],     t2 = Ts[mr * 16 + 4 + h];
        float t3 = Ts[mr * 16 + 8 + h], t4 = Ts[mr * 16 + 12 + h];
        float s12 = mk ? sigm_f(t1 * t2) : 0.f;   // sigmoid(-10000) == 0
        float s34 = mk ? sigm_f(t3 * t4) : 0.f;
        atomicAdd(&abuf[((size_t)(bb << 8) + j) * 4 + h], s12);  // sum over i
        atomicAdd(&bbuf[((size_t)(bb << 8) + i) * 4 + h], s34);  // sum over j
    }
}

// ---------------------------------------------------------------------------
// p += sigmoid(t5 * (og @ t6_W + t6_b)) with mask (masked: sigmoid(-1e4)=0).
// ---------------------------------------------------------------------------
__global__ void k_gate(float* __restrict__ P, const float* __restrict__ t5buf,
                       const float* __restrict__ abuf, const float* __restrict__ bbuf,
                       const float* __restrict__ t6W, const float* __restrict__ t6b,
                       const int* __restrict__ mask) {
    size_t idx = (size_t)blockIdx.x * 256 + threadIdx.x;
    int e = (int)(idx & 127);
    size_t r = idx >> 7;
    if (!mask[r]) return;
    int bb = (int)(r >> 16);
    int rem = (int)(r & 65535);
    int i = rem >> 8, j = rem & 255;
    const float* av = abuf + ((size_t)(bb << 8) + j) * 4;
    const float* bv = bbuf + ((size_t)(bb << 8) + i) * 4;
    float gate = t6b[e];
#pragma unroll
    for (int h = 0; h < 4; ++h) gate += (av[h] + bv[h]) * t6W[h * Ec + e];
    P[idx] += sigm_f(t5buf[idx] * gate);
}

// ---------------------------------------------------------------------------
// Fused pair FFN, 64 rows (4 M-tiles) per block:
// pn=LN(p,en3); h=gelu(pn@effn1+b) kept bf16 in LDS; p += h@effn2 + b.
// Each weight B fragment loaded once, reused by 4 WMMAs.
// LDS: 16 KB (As) + 64 KB (Hs) + ~1.5 KB scratch.
// ---------------------------------------------------------------------------
__global__ void k_effn(float* __restrict__ P, const float* __restrict__ gs,
                       const float* __restrict__ gb, const __bf16* __restrict__ e1sw,
                       const float* __restrict__ e1b, const __bf16* __restrict__ e2sw,
                       const float* __restrict__ e2b) {
    __shared__ __bf16 As[64 * Ec];     // 16 KB
    __shared__ __bf16 Hs[64 * EFc];    // 64 KB
    __shared__ float  red[256];
    __shared__ float  mArr[64], rArr[64];
    int mt = blockIdx.x, tid = threadIdx.x, lane = tid & 31, wave = tid >> 5;
    size_t base = (size_t)mt * 64 * Ec;
    tile_ln64(P, base, As, red, mArr, rArr, gs, gb, tid);

    const v16bf* W1 = (const v16bf*)e1sw;   // 128x512, 32 ntiles
    const v16bf* W2 = (const v16bf*)e2sw;   // 512x128, 8 ntiles
    int nlo = lane & 15;
    int mb  = (lane >> 4) << 3;

    // GEMM1: 128 -> 512, gelu, store bf16 into LDS
#pragma unroll
    for (int q = 0; q < 4; ++q) {
        int nt = wave * 4 + q;
        v8f c[4] = {};
#pragma unroll
        for (int kk = 0; kk < 4; ++kk) {
            v16bf b = W1[((size_t)kk * 32 + nt) * 32 + lane];
#pragma unroll
            for (int m = 0; m < 4; ++m) {
                v16bf a = load_a_frag(As + m * 16 * Ec, lane, kk, Ec);
                c[m] = wmma_bf16(a, b, c[m]);
            }
        }
        int n = (nt << 4) + nlo;
        float bv = e1b[n];
#pragma unroll
        for (int m = 0; m < 4; ++m)
#pragma unroll
            for (int r = 0; r < 8; ++r)
                Hs[(m * 16 + mb + r) * EFc + n] = (__bf16)gelu_f(c[m][r] + bv);
    }
    __syncthreads();

    // GEMM2: 512 -> 128, residual read from global, store to global
    v8f d[4] = {};
#pragma unroll
    for (int kk = 0; kk < 16; ++kk) {
        v16bf b = W2[((size_t)kk * 8 + wave) * 32 + lane];
#pragma unroll
        for (int m = 0; m < 4; ++m) {
            v16bf a = load_a_frag(Hs + m * 16 * EFc, lane, kk, EFc);
            d[m] = wmma_bf16(a, b, d[m]);
        }
    }
    int n = (wave << 4) + nlo;
    float bv = e2b[n];
#pragma unroll
    for (int m = 0; m < 4; ++m)
#pragma unroll
        for (int r = 0; r < 8; ++r) {
            size_t o = ((size_t)mt * 64 + m * 16 + mb + r) * Ec + n;
            P[o] = P[o] + d[m][r] + bv;
        }
}

// ---------------------------------------------------------------------------
// Host orchestration
// ---------------------------------------------------------------------------
extern "C" void kernel_launch(void* const* d_in, const int* in_sizes, int n_in,
                              void* d_out, int out_size, void* d_ws, size_t ws_size,
                              hipStream_t stream) {
    (void)in_sizes; (void)n_in; (void)out_size; (void)ws_size;
    const float* x       = (const float*)d_in[0];
    const float* p       = (const float*)d_in[1];
    const int*   mask    = (const int*)d_in[2];
    // d_in[3..6] = qkv/rqk (dead code)
    const float* nffn1_W = (const float*)d_in[7];
    const float* nffn1_b = (const float*)d_in[8];
    const float* nffn2_W = (const float*)d_in[9];
    const float* nffn2_b = (const float*)d_in[10];
    const float* o12_W   = (const float*)d_in[11];
    const float* o12_b   = (const float*)d_in[12];
    const float* o3_W    = (const float*)d_in[13];
    const float* o3_b    = (const float*)d_in[14];
    const float* t_W     = (const float*)d_in[15];
    const float* t_b     = (const float*)d_in[16];
    const float* t5_W    = (const float*)d_in[17];
    const float* t5_b    = (const float*)d_in[18];
    const float* t6_W    = (const float*)d_in[19];
    const float* t6_b    = (const float*)d_in[20];
    const float* effn1_W = (const float*)d_in[21];
    const float* effn1_b = (const float*)d_in[22];
    const float* effn2_W = (const float*)d_in[23];
    const float* effn2_b = (const float*)d_in[24];
    // d_in[25..26] = n1 (dead)
    const float* n2_s  = (const float*)d_in[27];
    const float* n2_b  = (const float*)d_in[28];
    const float* en1_s = (const float*)d_in[29];
    const float* en1_b = (const float*)d_in[30];
    const float* en2_s = (const float*)d_in[31];
    const float* en2_b = (const float*)d_in[32];
    const float* en3_s = (const float*)d_in[33];
    const float* en3_b = (const float*)d_in[34];

    float* xo = (float*)d_out;                 // (B,N,D)
    float* po = xo + (size_t)ROWS_X * Dc;      // (B,N,N,E)

    char* w = (char*)d_ws;
    auto carve = [&](size_t bytes) -> char* {
        char* r = w; w += (bytes + 255) & ~(size_t)255; return r;
    };
    __bf16* nffn1sw = (__bf16*)carve((size_t)Dc * NFc * 2);
    __bf16* nffn2sw = (__bf16*)carve((size_t)NFc * Dc * 2);
    __bf16* effn1sw = (__bf16*)carve((size_t)Ec * EFc * 2);
    __bf16* effn2sw = (__bf16*)carve((size_t)EFc * Ec * 2);
    __bf16* t5sw    = (__bf16*)carve((size_t)Ec * Ec * 2);
    __bf16* tsw     = (__bf16*)carve((size_t)Ec * 16 * 2);
    float*  xn2     = (float*)carve((size_t)ROWS_X * Dc * 4);
    float*  hbuf    = (float*)carve((size_t)ROWS_X * NFc * 4);
    float*  obuf    = (float*)carve((size_t)ROWS_X * 8 * 4);
    float*  abuf    = (float*)carve((size_t)Bc * Nc_ * Hc * 4);
    float*  bbuf    = (float*)carve((size_t)Bc * Nc_ * Hc * 4);
    float*  t5buf   = (float*)carve((size_t)ROWS_P * Ec * 4);

    auto swz = [&](const float* W, __bf16* o, int K, int N) {
        int total = (K >> 5) * (N >> 4) * 32;
        k_wswz<<<(total + 255) / 256, 256, 0, stream>>>(W, o, K, N);
    };
    // 0) one-time weight conversion to swizzled bf16
    swz(nffn1_W, nffn1sw, Dc, NFc);
    swz(nffn2_W, nffn2sw, NFc, Dc);
    swz(effn1_W, effn1sw, Ec, EFc);
    swz(effn2_W, effn2sw, EFc, Ec);
    swz(t5_W, t5sw, Ec, Ec);
    swz(t_W, tsw, Ec, 16);

    // 1) x-FFN: x_out = x + gelu(LN(x,n2) @ nffn1) @ nffn2
    k_ln<<<ROWS_X, 256, 0, stream>>>(x, n2_s, n2_b, xn2, Dc);
    k_gemm<<<dim3(ROWS_X / 16, NFc / 128), 256, 0, stream>>>(
        xn2, nffn1sw, nffn1_b, nullptr, hbuf, Dc, NFc, 1);
    k_gemm<<<dim3(ROWS_X / 16, Dc / 128), 256, 0, stream>>>(
        hbuf, nffn2sw, nffn2_b, x, xo, NFc, Dc, 0);

    // 2) outer-product pair update: p1 = p + (o1 (x) o2) @ o3_W + o3_b
    k_ln_o<<<ROWS_X, 256, 0, stream>>>(xo, en1_s, en1_b, o12_W, o12_b, obuf);
    k_pair1<<<(int)((size_t)ROWS_P * Ec / 256), 256, 0, stream>>>(
        p, obuf, o3_W, o3_b, po);

    // 3) triangle-gate: t5/tt WMMA (64-row blocks) + masked sigmoid reductions
    k_zero<<<(2 * Bc * Nc_ * Hc + 255) / 256, 256, 0, stream>>>(abuf, 2 * Bc * Nc_ * Hc);
    k_pair_t5<<<ROWS_P / 64, 256, 0, stream>>>(
        po, en2_s, en2_b, t5sw, t5_b, tsw, t_b, mask, t5buf, abuf, bbuf);
    k_gate<<<(int)((size_t)ROWS_P * Ec / 256), 256, 0, stream>>>(
        po, t5buf, abuf, bbuf, t6_W, t6_b, mask);

    // 4) pair FFN: p += gelu(LN(p,en3) @ effn1) @ effn2  (64-row blocks)
    k_effn<<<ROWS_P / 64, 256, 0, stream>>>(
        po, en3_s, en3_b, effn1sw, effn1_b, effn2sw, effn2_b);
}
